// PAM_Module_50646254355119
// MI455X (gfx1250) — matrix operand order repaired and checked by reading, not compiled, not run on hardware
//
#include <hip/hip_runtime.h>
#include <hip/hip_bf16.h>
#include <cstddef>
#include <cstdint>

// ---------------------------------------------------------------------------
// PAM module fused for MI455X (gfx1250, wave32, bf16 WMMA 16x16x32).
// prep (fold BN->bf16 weights) -> pool (pre-transposed) -> key GEMM (both
// layouts) -> fused q-GEMM / energy-GEMM / softmax / out-GEMM.
// Staging of bf16 tiles uses GLOBAL_LOAD_ASYNC_TO_LDS_B128 (ASYNCcnt).
// ---------------------------------------------------------------------------

typedef __attribute__((ext_vector_type(16))) __bf16 bfrag;   // A/B operand: 8 VGPRs
typedef __attribute__((ext_vector_type(8)))  float  v8f;     // C/D operand: 8 VGPRs
typedef __attribute__((ext_vector_type(2)))  __bf16 bf2;

#define LDW 72    // row stride (bf16) for 64-wide tiles (+8 pad -> 144 B rows)
#define LDQ 264   // row stride (bf16) for 256-wide tiles (+8 pad -> 528 B rows)
#define LDE 66    // row stride (f32) for 64-wide energy tile

__device__ __forceinline__ v8f wmma_bf(bfrag a, bfrag b, v8f c) {
  return __builtin_amdgcn_wmma_f32_16x16x32_bf16(false, a, false, b,
                                                 (short)0, c, false, false);
}

// A fragment (16x32 bf16) from row-major [M][K], row stride ld (ISA 7.12.2).
__device__ __forceinline__ bfrag load_fragA(const __bf16* __restrict__ t,
                                            int ld, int ks, int lane) {
  const int m = lane & 15, h = lane >> 4;
  const __bf16* p = t + m * ld + ks + 8 * h;
  bfrag f;
#pragma unroll
  for (int i = 0; i < 8; ++i) { f[i] = p[i]; f[i + 8] = p[16 + i]; }
  return f;
}

// B fragment (32x16 bf16) from transposed storage [N][K], row stride ld.
__device__ __forceinline__ bfrag load_fragB(const __bf16* __restrict__ t,
                                            int ld, int ks, int lane) {
  const int n = lane & 15, h = lane >> 4;
  const __bf16* p = t + n * ld + ks + 8 * h;
  bfrag f;
#pragma unroll
  for (int i = 0; i < 8; ++i) { f[i] = p[i]; f[i + 8] = p[16 + i]; }
  return f;
}

// Async global->LDS 16B copy (CDNA5, ASYNCcnt). GVS mode: saddr + 32b voffset.
__device__ __forceinline__ void async_b128(unsigned lds_byte, const void* base,
                                           unsigned byte_off) {
  asm volatile("global_load_async_to_lds_b128 %0, %1, %2"
               :
               : "v"(lds_byte), "v"(byte_off), "s"(base)
               : "memory");
}
__device__ __forceinline__ void async_wait0() {
  asm volatile("s_wait_asynccnt 0x0" ::: "memory");
}
__device__ __forceinline__ unsigned lds_addr(const void* p) {
  return (unsigned)(uintptr_t)p;   // low 32 bits of generic ptr = LDS offset
}

// ---------------------------------------------------------------------------
// prep: fold eval-mode BN scale into bf16 weights + per-channel fp32 bias.
// ---------------------------------------------------------------------------
__global__ void prep_kernel(const float* __restrict__ wq, const float* __restrict__ bq,
                            const float* __restrict__ gq, const float* __restrict__ betaq,
                            const float* __restrict__ mq, const float* __restrict__ vq,
                            const float* __restrict__ wk, const float* __restrict__ bk,
                            const float* __restrict__ gk, const float* __restrict__ betak,
                            const float* __restrict__ mk, const float* __restrict__ vk,
                            __bf16* __restrict__ wq_bf, float* __restrict__ biasq,
                            __bf16* __restrict__ wk_bf, float* __restrict__ biask) {
  int idx = blockIdx.x * blockDim.x + threadIdx.x;
  if (idx < 256 * 512) {
    int k = idx >> 9;
    float scq = gq[k] * rsqrtf(vq[k] + 1e-5f);
    float sck = gk[k] * rsqrtf(vk[k] + 1e-5f);
    wq_bf[idx] = (__bf16)(wq[idx] * scq);
    wk_bf[idx] = (__bf16)(wk[idx] * sck);
    if (idx < 256) {
      float sq = gq[idx] * rsqrtf(vq[idx] + 1e-5f);
      float sk = gk[idx] * rsqrtf(vk[idx] + 1e-5f);
      biasq[idx] = (bq[idx] - mq[idx]) * sq + betaq[idx];
      biask[idx] = (bk[idx] - mk[idx]) * sk + betak[idx];
    }
  }
}

// ---------------------------------------------------------------------------
// pool: AdaptiveAvgPool2d(8) -> bf16, stored TRANSPOSED as xp_t[b][p2=64][c=512]
// so the key kernel can async-copy K-contiguous rows straight into LDS.
// ---------------------------------------------------------------------------
__global__ void pool_kernel(const float* __restrict__ x, __bf16* __restrict__ xp_t) {
  int bc = blockIdx.x;            // b*512 + c
  int b = bc >> 9, c = bc & 511;
  int cell = threadIdx.x;         // 0..63 pooled cell
  int p = cell >> 3, q = cell & 7;
  const float* base = x + (size_t)bc * 4096 + p * 8 * 64 + q * 8;
  float s = 0.f;
#pragma unroll
  for (int i = 0; i < 8; ++i) {
    const float4* r = (const float4*)(base + i * 64);
    float4 a = r[0], b2 = r[1];
    s += a.x + a.y + a.z + a.w + b2.x + b2.y + b2.z + b2.w;
  }
  xp_t[((size_t)b * 64 + cell) * 512 + c] = (__bf16)(s * (1.f / 64.f));
}

// ---------------------------------------------------------------------------
// key: ReLU(Wk'[256,512] @ xp[b][512,64] + biask) -> key[b][256][64] (bf16)
//      and keyT[b][64][256] (bf16). 128 threads, 4 waves, 4Mx4N tiles/wave.
// ---------------------------------------------------------------------------
__global__ void __launch_bounds__(128)
key_kernel(const __bf16* __restrict__ xp_t, const __bf16* __restrict__ wk_bf,
           const float* __restrict__ biask, __bf16* __restrict__ key_bf,
           __bf16* __restrict__ keyT_bf) {
  extern __shared__ char smem[];
  __bf16* sW  = (__bf16*)(smem);                 // [256][LDW]
  __bf16* sXT = (__bf16*)(smem + 256 * LDW * 2); // [64][LDW]
  const unsigned sWu = lds_addr(sW), sXTu = lds_addr(sXT);

  const int b = blockIdx.x;
  const int tid = threadIdx.x, wv = tid >> 5, lane = tid & 31;
  const int h = lane >> 4, nl = lane & 15;

  v8f acc[4][4];
#pragma unroll
  for (int mi = 0; mi < 4; ++mi)
#pragma unroll
    for (int ni = 0; ni < 4; ++ni) { v8f z = {}; acc[mi][ni] = z; }

  for (int cc = 0; cc < 512; cc += 64) {
    for (int i = tid; i < 2048; i += 128) {        // sW: 256 rows x 8 b128
      int k = i >> 3, j = i & 7;
      async_b128(sWu + k * (LDW * 2) + j * 16, wk_bf,
                 (unsigned)((k * 512 + cc + j * 8) * 2));
    }
    for (int i = tid; i < 512; i += 128) {         // sXT: 64 rows x 8 b128
      int p = i >> 3, j = i & 7;
      async_b128(sXTu + p * (LDW * 2) + j * 16, xp_t,
                 (unsigned)(((b * 64 + p) * 512 + cc + j * 8) * 2));
    }
    async_wait0();
    __syncthreads();
#pragma unroll
    for (int ks = 0; ks < 64; ks += 32) {
      bfrag a[4];
#pragma unroll
      for (int mi = 0; mi < 4; ++mi)
        a[mi] = load_fragA(sW + (wv * 64 + mi * 16) * LDW, LDW, ks, lane);
#pragma unroll
      for (int ni = 0; ni < 4; ++ni) {
        bfrag bb = load_fragB(sXT + (ni * 16) * LDW, LDW, ks, lane);
#pragma unroll
        for (int mi = 0; mi < 4; ++mi)
          acc[mi][ni] = wmma_bf(a[mi], bb, acc[mi][ni]);
      }
    }
    __syncthreads();
  }
#pragma unroll
  for (int mi = 0; mi < 4; ++mi) {
    int ktile = (wv * 4 + mi) * 16;
#pragma unroll
    for (int ni = 0; ni < 4; ++ni) {
#pragma unroll
      for (int r = 0; r < 8; ++r) {
        int kidx = ktile + r + 8 * h;
        int pidx = ni * 16 + nl;
        float v = acc[mi][ni][r] + biask[kidx];
        v = v > 0.f ? v : 0.f;
        __bf16 bv = (__bf16)v;
        key_bf[((size_t)b * 256 + kidx) * 64 + pidx] = bv;
        keyT_bf[((size_t)b * 64 + pidx) * 256 + kidx] = bv;
      }
    }
  }
}

// ---------------------------------------------------------------------------
// main fused kernel: one WG = (batch b, 64-pixel tile). 128 threads, 4 waves.
// stage1: q = ReLU(Wq' x + biasq) -> sQ[n][k] bf16   (4Mx4N tiles per wave)
// stage2: energy[n][p] = q^T key  -> sE f32          (1Mx4N per wave)
// stage3: softmax rows            -> sAtt[n][p] bf16
// stage4: out = key @ att^T + q   -> global f32      (4Mx4N per wave)
// ---------------------------------------------------------------------------
__global__ void __launch_bounds__(128)
pam_main_kernel(const float* __restrict__ x, const __bf16* __restrict__ wq_bf,
                const float* __restrict__ biasq, const __bf16* __restrict__ key_bf,
                const __bf16* __restrict__ keyT_bf, float* __restrict__ out) {
  extern __shared__ char smem[];
  __bf16* sW   = (__bf16*)(smem + 0);       // [256][LDW]  36864 B (stage1)
  __bf16* sXT  = (__bf16*)(smem + 36864);   // [64][LDW]    9216 B (stage1)
  __bf16* sKT  = (__bf16*)(smem + 0);       // [64][LDQ]   33792 B (stage2+, alias)
  float*  sE   = (float*) (smem + 33792);   // [64][LDE]   16896 B (stage2+, alias)
  __bf16* sQ   = (__bf16*)(smem + 50688);   // [64][LDQ]   33792 B
  __bf16* sKA  = (__bf16*)(smem + 84480);   // [256][LDW]  36864 B
  __bf16* sAtt = (__bf16*)(smem + 121344);  // [64][LDW]    9216 B
  const unsigned sWu = lds_addr(sW), sKTu = lds_addr(sKT), sKAu = lds_addr(sKA);

  const int b  = blockIdx.x >> 6;
  const int n0 = (blockIdx.x & 63) * 64;
  const int tid = threadIdx.x, wv = tid >> 5, lane = tid & 31;
  const int h = lane >> 4, nl = lane & 15;

  v8f acc[4][4];
#pragma unroll
  for (int mi = 0; mi < 4; ++mi)
#pragma unroll
    for (int ni = 0; ni < 4; ++ni) { v8f z = {}; acc[mi][ni] = z; }

  // ---- stage 1: q projection, K-chunked 256x512x64 GEMM ----
  for (int cc = 0; cc < 512; cc += 64) {
    for (int i = tid; i < 2048; i += 128) {        // async: weight chunk
      int k = i >> 3, j = i & 7;
      async_b128(sWu + k * (LDW * 2) + j * 16, wq_bf,
                 (unsigned)((k * 512 + cc + j * 8) * 2));
    }
    for (int i = tid; i < 2048; i += 128) {        // manual: x fp32 -> bf16 pairs
      int n = i & 63, c0 = (i >> 6) * 2;
      float v0 = x[((size_t)(b * 512 + cc + c0)) * 4096 + n0 + n];
      float v1 = x[((size_t)(b * 512 + cc + c0 + 1)) * 4096 + n0 + n];
      bf2 t; t[0] = (__bf16)v0; t[1] = (__bf16)v1;
      *(bf2*)(sXT + n * LDW + c0) = t;
    }
    async_wait0();
    __syncthreads();
#pragma unroll
    for (int ks = 0; ks < 64; ks += 32) {
      bfrag a[4];
#pragma unroll
      for (int mi = 0; mi < 4; ++mi)
        a[mi] = load_fragA(sW + (wv * 64 + mi * 16) * LDW, LDW, ks, lane);
#pragma unroll
      for (int ni = 0; ni < 4; ++ni) {
        bfrag bb = load_fragB(sXT + (ni * 16) * LDW, LDW, ks, lane);
#pragma unroll
        for (int mi = 0; mi < 4; ++mi)
          acc[mi][ni] = wmma_bf(a[mi], bb, acc[mi][ni]);
      }
    }
    __syncthreads();
  }
  // epilogue: bias + ReLU, store q transposed for stage 2 (A layout)
#pragma unroll
  for (int mi = 0; mi < 4; ++mi) {
    int ktile = (wv * 4 + mi) * 16;
#pragma unroll
    for (int ni = 0; ni < 4; ++ni) {
#pragma unroll
      for (int r = 0; r < 8; ++r) {
        int kidx = ktile + r + 8 * h;
        float v = acc[mi][ni][r] + biasq[kidx];
        v = v > 0.f ? v : 0.f;
        sQ[(ni * 16 + nl) * LDQ + kidx] = (__bf16)v;
      }
    }
  }
  __syncthreads();

  // ---- async-load key in both layouts (sW/sXT regions now dead) ----
  for (int i = tid; i < 2048; i += 128) {          // sKA [k][p]
    int k = i >> 3, j = i & 7;
    async_b128(sKAu + k * (LDW * 2) + j * 16, key_bf,
               (unsigned)(((b * 256 + k) * 64 + j * 8) * 2));
  }
  for (int i = tid; i < 2048; i += 128) {          // sKT [p][k]
    int p = i >> 5, j = i & 31;
    async_b128(sKTu + p * (LDQ * 2) + j * 16, keyT_bf,
               (unsigned)(((b * 64 + p) * 256 + j * 8) * 2));
  }
  async_wait0();
  __syncthreads();

  // ---- stage 2: energy[n][p] = q^T @ key (wave wv owns n-tile wv) ----
  {
    v8f e[4];
#pragma unroll
    for (int ni = 0; ni < 4; ++ni) { v8f z = {}; e[ni] = z; }
    for (int kc = 0; kc < 256; kc += 32) {
      bfrag aq = load_fragA(sQ + (wv * 16) * LDQ, LDQ, kc, lane);
#pragma unroll
      for (int ni = 0; ni < 4; ++ni) {
        bfrag bk = load_fragB(sKT + (ni * 16) * LDQ, LDQ, kc, lane);
        e[ni] = wmma_bf(aq, bk, e[ni]);
      }
    }
#pragma unroll
    for (int ni = 0; ni < 4; ++ni)
#pragma unroll
      for (int r = 0; r < 8; ++r)
        sE[(wv * 16 + r + 8 * h) * LDE + ni * 16 + nl] = e[ni][r];
  }
  __syncthreads();

  // ---- stage 3: fp32 softmax over p, write att in B layout [n][p] ----
  if (tid < 64) {
    float mx = -3.402823e38f;
    for (int p = 0; p < 64; ++p) mx = fmaxf(mx, sE[tid * LDE + p]);
    float s = 0.f;
    for (int p = 0; p < 64; ++p) s += __expf(sE[tid * LDE + p] - mx);
    float inv = 1.f / s;
    for (int p = 0; p < 64; ++p)
      sAtt[tid * LDW + p] = (__bf16)(__expf(sE[tid * LDE + p] - mx) * inv);
  }
  __syncthreads();

  // ---- stage 4: out[k][n] = key @ att^T + q ----
#pragma unroll
  for (int mi = 0; mi < 4; ++mi)
#pragma unroll
    for (int ni = 0; ni < 4; ++ni) { v8f z = {}; acc[mi][ni] = z; }
#pragma unroll
  for (int kc = 0; kc < 64; kc += 32) {
    bfrag a[4];
#pragma unroll
    for (int mi = 0; mi < 4; ++mi)
      a[mi] = load_fragA(sKA + (wv * 64 + mi * 16) * LDW, LDW, kc, lane);
#pragma unroll
    for (int ni = 0; ni < 4; ++ni) {
      bfrag bb = load_fragB(sAtt + (ni * 16) * LDW, LDW, kc, lane);
#pragma unroll
      for (int mi = 0; mi < 4; ++mi)
        acc[mi][ni] = wmma_bf(a[mi], bb, acc[mi][ni]);
    }
  }
#pragma unroll
  for (int mi = 0; mi < 4; ++mi) {
    int ktile = (wv * 4 + mi) * 16;
#pragma unroll
    for (int ni = 0; ni < 4; ++ni) {
#pragma unroll
      for (int r = 0; r < 8; ++r) {
        int kidx = ktile + r + 8 * h;
        int nidx = ni * 16 + nl;
        float q0 = (float)sQ[nidx * LDQ + kidx];
        out[((size_t)(b * 256 + kidx)) * 4096 + n0 + nidx] = acc[mi][ni][r] + q0;
      }
    }
  }
}

// ---------------------------------------------------------------------------
// host launcher
// ---------------------------------------------------------------------------
extern "C" void kernel_launch(void* const* d_in, const int* in_sizes, int n_in,
                              void* d_out, int out_size, void* d_ws, size_t ws_size,
                              hipStream_t stream) {
  (void)in_sizes; (void)n_in; (void)out_size; (void)ws_size;
  const float* x     = (const float*)d_in[0];
  const float* wq    = (const float*)d_in[1];
  const float* bq    = (const float*)d_in[2];
  const float* gq    = (const float*)d_in[3];
  const float* betaq = (const float*)d_in[4];
  const float* mq    = (const float*)d_in[5];
  const float* vq    = (const float*)d_in[6];
  const float* wk    = (const float*)d_in[7];
  const float* bk    = (const float*)d_in[8];
  const float* gk    = (const float*)d_in[9];
  const float* betak = (const float*)d_in[10];
  const float* mk    = (const float*)d_in[11];
  const float* vk    = (const float*)d_in[12];
  float* out = (float*)d_out;

  char* ws = (char*)d_ws;
  __bf16* wq_bf   = (__bf16*)(ws + 0);        // 262144
  __bf16* wk_bf   = (__bf16*)(ws + 262144);   // 262144
  float*  biasq   = (float*) (ws + 524288);   // 1024
  float*  biask   = (float*) (ws + 525312);   // 1024
  __bf16* xp_t    = (__bf16*)(ws + 526336);   // 16*64*512*2 = 1048576
  __bf16* key_bf  = (__bf16*)(ws + 1574912);  // 16*256*64*2 = 524288
  __bf16* keyT_bf = (__bf16*)(ws + 2099200);  // 16*64*256*2 = 524288
  // total workspace used: 2623488 bytes

  prep_kernel<<<512, 256, 0, stream>>>(wq, bq, gq, betaq, mq, vq,
                                       wk, bk, gk, betak, mk, vk,
                                       wq_bf, biasq, wk_bf, biask);
  pool_kernel<<<16 * 512, 64, 0, stream>>>(x, xp_t);
  key_kernel<<<16, 128, (256 * LDW + 64 * LDW) * 2, stream>>>(
      xp_t, wk_bf, biask, key_bf, keyT_bf);
  pam_main_kernel<<<16 * 64, 128, 130560, stream>>>(
      x, wq_bf, biasq, key_bf, keyT_bf, out);
}
